// TbNetV2_5196910429029
// MI455X (gfx1250) — matrix operand order repaired and checked by reading, not compiled, not run on hardware
//
#include <hip/hip_runtime.h>
#include <hip/hip_bf16.h>

typedef __attribute__((ext_vector_type(16))) __bf16 v16bf;
typedef __attribute__((ext_vector_type(8)))  float  v8f;
typedef unsigned int u32x4 __attribute__((ext_vector_type(4)));
typedef int          i32x4 __attribute__((ext_vector_type(4)));
typedef int          i32x8 __attribute__((ext_vector_type(8)));

typedef __attribute__((address_space(3))) const unsigned short* lds_cus_p;
typedef __attribute__((address_space(3))) const u32x4*          lds_cuv4_p;

union Frag {
    unsigned int u[8];
    v16bf v;
};

__device__ __forceinline__ unsigned short f32_to_bf16(float f) {
    unsigned int u = __float_as_uint(f);
    u += 0x7FFFu + ((u >> 16) & 1u);   // round-to-nearest-even
    return (unsigned short)(u >> 16);
}
__device__ __forceinline__ float bf16_to_f32(unsigned short h) {
    return __uint_as_float(((unsigned int)h) << 16);
}

enum AMode { A_DENSE_F32 = 0, A_DENSE_BF16 = 1, A_GATHER1 = 2, A_GATHER2 = 3 };

// ---------------------------------------------------------------------------
// Staged bf16 WMMA GEMM:  C[M,Nout] = op(A)[M,K] @ W[K,Nout] (+bias)(+relu)
// Weights bf16, layout [Nout, Kw] (Kw = K padded to 32).
// Block: 256 threads = 8 waves; 16 rows per block; wave w owns N-tiles
// w, w+8, w+16 (NTPW per wave).  CONTRACT: Nout == NTPW * 128.
// ---------------------------------------------------------------------------
template <int NTPW, int MODE>
__global__ __launch_bounds__(256) void wmma_gemm(
    int M, int K, int Kw,
    const float* __restrict__ Af, int lda,
    const int* __restrict__ idx0, const int* __restrict__ idx1, int idxStride, int featW,
    const unsigned short* __restrict__ WbT, const float* __restrict__ bias, int relu,
    float* __restrict__ Cf, unsigned short* __restrict__ Cb, int ldc)
{
    __shared__ unsigned short As[16 * 40];   // 16 rows x 32 k, stride 40 (16B-aligned rows)
    __shared__ int Is[16], Id[16];
    const int tid  = threadIdx.x;
    const int lane = tid & 31;
    const int wave = tid >> 5;
    const int m0   = blockIdx.x * 16;
    const int hi   = lane >> 4;   // 0/1 half-wave
    const int nlo  = lane & 15;

    if (MODE == A_GATHER1 || MODE == A_GATHER2) {
        if (tid < 16) {
            int grow = m0 + tid;
            int i0v = 0, i1v = 0;
            if (grow < M) {
                i0v = idx0[(long)grow * idxStride];
                if (MODE == A_GATHER2) i1v = idx1[grow];
            }
            Is[tid] = i0v;
            Id[tid] = i1v;
        }
        __syncthreads();
    }

    v8f acc[NTPW];
#pragma unroll
    for (int t = 0; t < NTPW; ++t)
#pragma unroll
        for (int j = 0; j < 8; ++j) acc[t][j] = 0.0f;

    for (int k0 = 0; k0 < K; k0 += 32) {
        // ---- stage 16x32 bf16 chunk of A into LDS (zero padded) ----
#pragma unroll
        for (int i = 0; i < 2; ++i) {
            int e  = tid * 2 + i;        // 0..511
            int r  = e >> 5;
            int kk = e & 31;
            int gk = k0 + kk;
            int grow = m0 + r;
            unsigned short val = 0;
            if (grow < M && gk < K) {
                float f;
                if (MODE == A_DENSE_F32) {
                    f = Af[(long)grow * lda + gk];
                } else if (MODE == A_GATHER1) {
                    f = Af[(long)Is[r] * featW + gk];
                } else { // A_GATHER2: concat(feat[src], feat[dst])
                    f = (gk < featW) ? Af[(long)Is[r] * featW + gk]
                                     : Af[(long)Id[r] * featW + (gk - featW)];
                }
                val = f32_to_bf16(f);
            }
            As[r * 40 + kk] = val;
        }
        __syncthreads();

        // ---- A fragment (16-bit A 16x32 layout), row m = nlo ----
        Frag a;
        {
            const u32x4 a0 = *(const u32x4*)&As[nlo * 40 + hi * 16];
            const u32x4 a1 = *(const u32x4*)&As[nlo * 40 + hi * 16 + 8];
            a.u[0] = a0.x; a.u[1] = a0.y; a.u[2] = a0.z; a.u[3] = a0.w;
            a.u[4] = a1.x; a.u[5] = a1.y; a.u[6] = a1.z; a.u[7] = a1.w;
        }

#pragma unroll
        for (int t = 0; t < NTPW; ++t) {
            int n = ((wave + t * 8) << 4) + nlo;
            // B fragment: 32x16 16-bit layout; lane holds column n.
            const unsigned short* wrow = WbT + (long)n * Kw + k0 + hi * 16;
            Frag b;
            const u32x4 b0 = *(const u32x4*)wrow;
            const u32x4 b1 = *(const u32x4*)(wrow + 8);
            b.u[0] = b0.x; b.u[1] = b0.y; b.u[2] = b0.z; b.u[3] = b0.w;
            b.u[4] = b1.x; b.u[5] = b1.y; b.u[6] = b1.z; b.u[7] = b1.w;
            acc[t] = __builtin_amdgcn_wmma_f32_16x16x32_bf16(
                false, a.v, false, b.v, (short)0, acc[t], false, false);
        }
        __syncthreads();
    }

    // ---- epilogue: f32 C/D 16x16 layout ----
#pragma unroll
    for (int t = 0; t < NTPW; ++t) {
        int ncol = ((wave + t * 8) << 4) + nlo;
        float bv = bias ? bias[ncol] : 0.0f;
#pragma unroll
        for (int v = 0; v < 8; ++v) {
            int row = m0 + hi * 8 + v;
            if (row < M) {
                float val = acc[t][v] + bv;
                if (relu) val = fmaxf(val, 0.0f);
                if (Cf) Cf[(long)row * ldc + ncol] = val;
                else    Cb[(long)row * ldc + ncol] = f32_to_bf16(val);
            }
        }
    }
}

// ---------------------------------------------------------------------------
// TDM-based GEMM for dense bf16 A (lda == K, K <= 384, Nout == 128):
// one tensor_load_to_lds pulls the whole 16 x K bf16 A-tile into LDS
// (rows beyond M read zero via tensor_dim1), then 8 waves run all K-chunks
// with no further staging barriers.  The LDS pointer is laundered IN
// ADDRESS SPACE 3 through an asm memory-clobber: opaque to the optimizer
// (so the TDM-written LDS reads cannot fold to undef) while keeping the
// LDS-ness so the reads lower to ds_load_b128 (not flat_load).  Two
// accumulators break the WMMA->WMMA C-chain hazard.
// ---------------------------------------------------------------------------
__global__ __launch_bounds__(256) void tdm_gemm(
    int M, int K, int Kw,
    const unsigned short* __restrict__ Ab,
    const unsigned short* __restrict__ WbT,
    const float* __restrict__ bias, int relu,
    float* __restrict__ Cf, unsigned short* __restrict__ Cb, int ldc)
{
    __shared__ unsigned short As[16 * 384];  // only LDS object => lds offset 0
    const int tid  = threadIdx.x;
    const int lane = tid & 31;
    const int wave = tid >> 5;
    const int m0   = blockIdx.x * 16;
    const int hi   = lane >> 4;
    const int nlo  = lane & 15;

    if (wave == 0) {
        unsigned long long ga =
            (unsigned long long)Ab + (unsigned long long)m0 * (unsigned)K * 2ull;
        unsigned int rows = (unsigned int)(M - m0);      // tensor_dim1 (OOB rows -> 0)
        unsigned int uK   = (unsigned int)K;
        u32x4 g0 = {
            1u,                                          // count=1, user descriptor
            0u,                                          // lds_addr = 0
            (unsigned int)(ga & 0xFFFFFFFFull),
            (unsigned int)((ga >> 32) & 0x01FFFFFFull) | 0x80000000u  // type=2
        };
        i32x8 g1 = {
            (int)0x10000u,                               // data_size=1 (2 bytes)
            (int)((uK & 0xFFFFu) << 16),                 // tensor_dim0[15:0]
            (int)(((uK >> 16) & 0xFFFFu) | ((rows & 0xFFFFu) << 16)),   // td0 hi | td1 lo
            (int)(((rows >> 16) & 0xFFFFu) | ((uK & 0xFFFFu) << 16)),   // td1 hi | tile_dim0=K
            (int)16,                                     // tile_dim1=16, tile_dim2=0
            (int)uK,                                     // tensor_dim0_stride[31:0]
            0, 0
        };
        i32x4 gz = {0, 0, 0, 0};
#if __clang_major__ >= 23
        i32x8 gz8 = {0, 0, 0, 0, 0, 0, 0, 0};
        __builtin_amdgcn_tensor_load_to_lds(g0, g1, gz, gz, gz8, 0);
#else
        __builtin_amdgcn_tensor_load_to_lds(g0, g1, gz, gz, 0);
#endif
        __builtin_amdgcn_s_wait_tensorcnt(0);
    }
    __syncthreads();

    // Launder the LDS pointer in addrspace(3): opaque to the optimizer,
    // ordered after the TDM completion + barrier, still lowers to ds_load.
    lds_cus_p asp = (lds_cus_p)&As[0];
    asm volatile("" : "+v"(asp) : : "memory");

    v8f acc0, acc1;
#pragma unroll
    for (int j = 0; j < 8; ++j) { acc0[j] = 0.0f; acc1[j] = 0.0f; }

    const int n = (wave << 4) + nlo;
    const int arow = nlo * K + hi * 16;
    for (int k0 = 0; k0 < K; k0 += 64) {
        {   // even 32-chunk
            Frag a, b;
            const u32x4 a0 = *(lds_cuv4_p)&asp[arow + k0];
            const u32x4 a1 = *(lds_cuv4_p)&asp[arow + k0 + 8];
            a.u[0] = a0.x; a.u[1] = a0.y; a.u[2] = a0.z; a.u[3] = a0.w;
            a.u[4] = a1.x; a.u[5] = a1.y; a.u[6] = a1.z; a.u[7] = a1.w;
            const unsigned short* wrow = WbT + (long)n * Kw + k0 + hi * 16;
            if (k0 + 64 < K) __builtin_prefetch(wrow + 64, 0, 1);
            const u32x4 b0 = *(const u32x4*)wrow;
            const u32x4 b1 = *(const u32x4*)(wrow + 8);
            b.u[0] = b0.x; b.u[1] = b0.y; b.u[2] = b0.z; b.u[3] = b0.w;
            b.u[4] = b1.x; b.u[5] = b1.y; b.u[6] = b1.z; b.u[7] = b1.w;
            acc0 = __builtin_amdgcn_wmma_f32_16x16x32_bf16(
                false, a.v, false, b.v, (short)0, acc0, false, false);
        }
        if (k0 + 32 < K) {   // odd 32-chunk (independent accumulator)
            Frag a, b;
            const u32x4 a0 = *(lds_cuv4_p)&asp[arow + k0 + 32];
            const u32x4 a1 = *(lds_cuv4_p)&asp[arow + k0 + 40];
            a.u[0] = a0.x; a.u[1] = a0.y; a.u[2] = a0.z; a.u[3] = a0.w;
            a.u[4] = a1.x; a.u[5] = a1.y; a.u[6] = a1.z; a.u[7] = a1.w;
            const unsigned short* wrow = WbT + (long)n * Kw + k0 + 32 + hi * 16;
            const u32x4 b0 = *(const u32x4*)wrow;
            const u32x4 b1 = *(const u32x4*)(wrow + 8);
            b.u[0] = b0.x; b.u[1] = b0.y; b.u[2] = b0.z; b.u[3] = b0.w;
            b.u[4] = b1.x; b.u[5] = b1.y; b.u[6] = b1.z; b.u[7] = b1.w;
            acc1 = __builtin_amdgcn_wmma_f32_16x16x32_bf16(
                false, a.v, false, b.v, (short)0, acc1, false, false);
        }
    }

    float bv = bias ? bias[n] : 0.0f;
#pragma unroll
    for (int v = 0; v < 8; ++v) {
        int row = m0 + hi * 8 + v;
        if (row < M) {
            float val = acc0[v] + acc1[v] + bv;
            if (relu) val = fmaxf(val, 0.0f);
            if (Cf) Cf[(long)row * ldc + n] = val;
            else    Cb[(long)row * ldc + n] = f32_to_bf16(val);
        }
    }
}

// ---------------------------------------------------------------------------
// Small helper kernels
// ---------------------------------------------------------------------------
__global__ void fill_f32(float* p, float v, long n) {
    long i = (long)blockIdx.x * blockDim.x + threadIdx.x;
    if (i < n) p[i] = v;
}

// dst layout [N, Kw] bf16; src either [K,N] (srcIsKN=1) or [N,K] (srcIsKN=0)
__global__ void convert_weight(const float* __restrict__ src, unsigned short* __restrict__ dst,
                               int K, int N, int Kw, int srcIsKN) {
    long i = (long)blockIdx.x * blockDim.x + threadIdx.x;
    if (i >= (long)N * Kw) return;
    int n = (int)(i / Kw), k = (int)(i % Kw);
    float v = 0.0f;
    if (k < K) v = srcIsKN ? src[(long)k * N + n] : src[(long)n * K + k];
    dst[i] = f32_to_bf16(v);
}

__global__ void deg_atomic(const int* __restrict__ dst, float* deg, int E) {
    int e = blockIdx.x * blockDim.x + threadIdx.x;
    if (e < E) atomicAdd(&deg[dst[e]], 1.0f);
}

__global__ void deg_rsqrt(const float* __restrict__ deg, float* dinv, int N) {
    int i = blockIdx.x * blockDim.x + threadIdx.x;
    if (i < N) dinv[i] = rsqrtf(deg[i]);
}

// agg = bias + xw * dinv^2   (self-loop contribution)
__global__ void agg_init(const float* __restrict__ xw, const float* __restrict__ dinv,
                         const float* __restrict__ bias, float* agg, int N, int Hd) {
    long t = (long)blockIdx.x * blockDim.x + threadIdx.x;
    if (t >= (long)N * Hd) return;
    int i = (int)(t / Hd), f = (int)(t % Hd);
    float di = dinv[i];
    agg[t] = bias[f] + xw[t] * di * di;
}

// agg[dst] += xw[src] * dinv[src]*dinv[dst]
__global__ void gcn_scatter(const float* __restrict__ xw, const int* __restrict__ src,
                            const int* __restrict__ dst, const float* __restrict__ dinv,
                            float* agg, int E, int Hd) {
    long gid = (long)blockIdx.x * blockDim.x + threadIdx.x;
    int chunks = Hd >> 5;
    if (gid >= (long)E * chunks) return;
    int e = (int)(gid / chunks), c = (int)(gid % chunks);
    int s = src[e], d = dst[e];
    float w = dinv[s] * dinv[d];
    const float* xr = xw + (long)s * Hd + c * 32;
    float* ar = agg + (long)d * Hd + c * 32;
#pragma unroll 4
    for (int j = 0; j < 32; ++j) atomicAdd(&ar[j], xr[j] * w);
}

__global__ void relu_copy(const float* __restrict__ src, float* dst, long n) {
    long i = (long)blockIdx.x * blockDim.x + threadIdx.x;
    if (i < n) dst[i] = fmaxf(src[i], 0.0f);
}

// GRU gate fusion: biases already folded into gi/gh by the GEMMs.
__global__ void gru_gate(const float* __restrict__ gi, const float* __restrict__ gh,
                         float* h, float* ts, int N, int Hd) {
    long t = (long)blockIdx.x * blockDim.x + threadIdx.x;
    if (t >= (long)N * Hd) return;
    int n = (int)(t / Hd), j = (int)(t % Hd);
    long o = (long)n * 3 * Hd;
    float r = 1.0f / (1.0f + expf(-(gi[o + j] + gh[o + j])));
    float z = 1.0f / (1.0f + expf(-(gi[o + Hd + j] + gh[o + Hd + j])));
    float c = tanhf(gi[o + 2 * Hd + j] + r * gh[o + 2 * Hd + j]);
    float hn = (1.0f - z) * c + z * h[t];
    h[t] = hn;
    ts[t] += hn;
}

// hidden[E,128] bf16 @ W2[128,2] + b2 -> log_softmax -> out[E,2]
__global__ void head_kernel(const unsigned short* __restrict__ hid,
                            const float* __restrict__ W2, const float* __restrict__ b2,
                            float* out, int E, int Hd) {
    int e = blockIdx.x * blockDim.x + threadIdx.x;
    if (e >= E) return;
    float a0 = b2[0], a1 = b2[1];
    const unsigned short* hr = hid + (long)e * Hd;
    for (int j = 0; j < Hd; ++j) {
        float hv = bf16_to_f32(hr[j]);
        a0 += hv * W2[j * 2 + 0];
        a1 += hv * W2[j * 2 + 1];
    }
    float m = fmaxf(a0, a1);
    float ls = logf(expf(a0 - m) + expf(a1 - m));
    out[(long)e * 2 + 0] = a0 - m - ls;
    out[(long)e * 2 + 1] = a1 - m - ls;
}

// ---------------------------------------------------------------------------
// Host orchestration
// ---------------------------------------------------------------------------
static inline long cdivl(long a, long b) { return (a + b - 1) / b; }
static inline int  kpad32(int k) { return (k + 31) & ~31; }

extern "C" void kernel_launch(void* const* d_in, const int* in_sizes, int n_in,
                              void* d_out, int out_size, void* d_ws, size_t ws_size,
                              hipStream_t stream) {
    const float* x              = (const float*)d_in[0];
    const int*   edge_index     = (const int*)d_in[1];
    const int*   xtext          = (const int*)d_in[2];
    const float* image_features = (const float*)d_in[3];
    const float* gcn1_W  = (const float*)d_in[4];  const float* gcn1_b  = (const float*)d_in[5];
    const float* gcn2_W  = (const float*)d_in[6];  const float* gcn2_b  = (const float*)d_in[7];
    const float* gcnt1_W = (const float*)d_in[8];  const float* gcnt1_b = (const float*)d_in[9];
    const float* gcnt2_W = (const float*)d_in[10]; const float* gcnt2_b = (const float*)d_in[11];
    const float* embed   = (const float*)d_in[12];
    const float* gru_Wih = (const float*)d_in[13];
    const float* gru_Whh = (const float*)d_in[14];
    const float* gru_bih = (const float*)d_in[15];
    const float* gru_bhh = (const float*)d_in[16];
    const float* lin_pos_W  = (const float*)d_in[17]; const float* lin_pos_b  = (const float*)d_in[18];
    const float* lin_text_W = (const float*)d_in[19]; const float* lin_text_b = (const float*)d_in[20];
    const float* lin_img_W  = (const float*)d_in[21]; const float* lin_img_b  = (const float*)d_in[22];
    const float* lin_row_W1 = (const float*)d_in[23]; const float* lin_row_b1 = (const float*)d_in[24];
    const float* lin_row_W2 = (const float*)d_in[25]; const float* lin_row_b2 = (const float*)d_in[26];
    const float* lin_col_W1 = (const float*)d_in[27]; const float* lin_col_b1 = (const float*)d_in[28];
    const float* lin_col_W2 = (const float*)d_in[29]; const float* lin_col_b2 = (const float*)d_in[30];

    const int H    = in_sizes[5];               // 128
    const int FIN  = in_sizes[4] / H;           // 8
    const int N    = in_sizes[0] / FIN;         // 20000
    const int E    = in_sizes[1] / 2;           // 320000
    const int T    = in_sizes[2] / N;           // 16
    const int H3   = 3 * H;                     // 384
    const int ETXT = in_sizes[13] / H3;         // 64
    const int IMGW = in_sizes[3] / N;           // 256 (= H*NS)

    const int* srcIdx = edge_index;
    const int* dstIdx = edge_index + E;

    // ---- carve workspace ----
    char* wp = (char*)d_ws;
    auto alloc = [&](size_t bytes) {
        char* p = wp;
        wp += (bytes + 255) & ~(size_t)255;
        return p;
    };
    float* deg     = (float*)alloc((size_t)N * 4);
    float* dinv    = (float*)alloc((size_t)N * 4);
    float* xw      = (float*)alloc((size_t)N * H * 4);
    float* tbuf    = (float*)alloc((size_t)N * H * 4);
    float* pos     = (float*)alloc((size_t)N * H * 4);
    float* text    = (float*)alloc((size_t)N * H * 4);
    float* hstate  = (float*)alloc((size_t)N * H * 4);
    float* textsum = (float*)alloc((size_t)N * H * 4);
    float* gi      = (float*)alloc((size_t)N * H3 * 4);
    float* gh      = (float*)alloc((size_t)N * H3 * 4);
    unsigned short* ef  = (unsigned short*)alloc((size_t)E * H3 * 2);   // [E, 384] bf16
    unsigned short* hid = (unsigned short*)alloc((size_t)E * H * 2);    // [E, 128] bf16
    // bf16 weights, layout [N, Kw]
    unsigned short* wb_gcn1  = (unsigned short*)alloc((size_t)H  * kpad32(FIN) * 2);
    unsigned short* wb_gcn2  = (unsigned short*)alloc((size_t)H  * H * 2);
    unsigned short* wb_gcnt1 = (unsigned short*)alloc((size_t)H  * H * 2);
    unsigned short* wb_gcnt2 = (unsigned short*)alloc((size_t)H  * H * 2);
    unsigned short* wb_wih   = (unsigned short*)alloc((size_t)H3 * ETXT * 2);
    unsigned short* wb_whh   = (unsigned short*)alloc((size_t)H3 * H * 2);
    unsigned short* wb_pos   = (unsigned short*)alloc((size_t)H  * 2 * H * 2);
    unsigned short* wb_text  = (unsigned short*)alloc((size_t)H  * 2 * H * 2);
    unsigned short* wb_img   = (unsigned short*)alloc((size_t)H  * 2 * IMGW * 2);
    unsigned short* wb_row1  = (unsigned short*)alloc((size_t)H  * H3 * 2);
    unsigned short* wb_col1  = (unsigned short*)alloc((size_t)H  * H3 * 2);

    float* out_row = (float*)d_out;
    float* out_col = out_row + (long)E * 2;

    const int TB = 256;
    auto cwt = [&](const float* s, unsigned short* d, int K, int Nn, int srcIsKN) {
        int Kw = kpad32(K);
        long n = (long)Nn * Kw;
        convert_weight<<<(unsigned)cdivl(n, TB), TB, 0, stream>>>(s, d, K, Nn, Kw, srcIsKN);
    };
    // weights stored [in,out] -> srcIsKN=1; GRU weights stored [out,in] -> 0
    cwt(gcn1_W,  wb_gcn1,  FIN,      H,  1);
    cwt(gcn2_W,  wb_gcn2,  H,        H,  1);
    cwt(gcnt1_W, wb_gcnt1, H,        H,  1);
    cwt(gcnt2_W, wb_gcnt2, H,        H,  1);
    cwt(gru_Wih, wb_wih,   ETXT,     H3, 0);
    cwt(gru_Whh, wb_whh,   H,        H3, 0);
    cwt(lin_pos_W,  wb_pos,  2 * H,    H, 1);
    cwt(lin_text_W, wb_text, 2 * H,    H, 1);
    cwt(lin_img_W,  wb_img,  2 * IMGW, H, 1);
    cwt(lin_row_W1, wb_row1, H3,       H, 1);
    cwt(lin_col_W1, wb_col1, H3,       H, 1);

    // amode-dispatched GEMM launches (contract: Nout == NTPW*128)
    auto gemm_f32 = [&](int M, int K, int Nout, const float* Af, int lda,
                        const unsigned short* W, const float* bias, int relu,
                        float* Cf, unsigned short* Cb, int ldc) {
        int Kw = kpad32(K);
        unsigned grid = (unsigned)cdivl(M, 16);
        if (Nout == 128)
            wmma_gemm<1, A_DENSE_F32><<<grid, 256, 0, stream>>>(
                M, K, Kw, Af, lda, nullptr, nullptr, 0, 0, W, bias, relu, Cf, Cb, ldc);
        else
            wmma_gemm<3, A_DENSE_F32><<<grid, 256, 0, stream>>>(
                M, K, Kw, Af, lda, nullptr, nullptr, 0, 0, W, bias, relu, Cf, Cb, ldc);
    };
    auto gemm_gather1 = [&](int M, int K, const float* feat, const int* idx, int istr,
                            const unsigned short* W, const float* bias,
                            float* Cf, int ldc) {   // Nout = 384
        int Kw = kpad32(K);
        unsigned grid = (unsigned)cdivl(M, 16);
        wmma_gemm<3, A_GATHER1><<<grid, 256, 0, stream>>>(
            M, K, Kw, feat, 0, idx, nullptr, istr, K, W, bias, 0, Cf, nullptr, ldc);
    };
    auto gemm_gather2 = [&](int M, int K, const float* feat, int featW,
                            const unsigned short* W, const float* bias,
                            unsigned short* Cb, int ldc) {   // Nout = 128, relu
        int Kw = kpad32(K);
        unsigned grid = (unsigned)cdivl(M, 16);
        wmma_gemm<1, A_GATHER2><<<grid, 256, 0, stream>>>(
            M, K, Kw, feat, 0, srcIdx, dstIdx, 1, featW, W, bias, 1, nullptr, Cb, ldc);
    };
    auto gemm_tdm = [&](int M, int K, const unsigned short* Ab,
                        const unsigned short* W, const float* bias,
                        unsigned short* Cb, int ldc) {   // Nout = 128, relu, lda==K
        int Kw = kpad32(K);
        unsigned grid = (unsigned)cdivl(M, 16);
        tdm_gemm<<<grid, 256, 0, stream>>>(M, K, Kw, Ab, W, bias, 1, nullptr, Cb, ldc);
    };

    // ---- GCN normalization (self-loops => deg starts at 1) ----
    fill_f32<<<(unsigned)cdivl(N, TB), TB, 0, stream>>>(deg, 1.0f, N);
    deg_atomic<<<(unsigned)cdivl(E, TB), TB, 0, stream>>>(dstIdx, deg, E);
    deg_rsqrt<<<(unsigned)cdivl(N, TB), TB, 0, stream>>>(deg, dinv, N);

    long NH = (long)N * H;
    auto gcn_layer = [&](const float* hin, int K, const unsigned short* Wb,
                         const float* bias, float* hout) {
        gemm_f32(N, K, H, hin, K, Wb, nullptr, 0, xw, nullptr, H);
        agg_init<<<(unsigned)cdivl(NH, TB), TB, 0, stream>>>(xw, dinv, bias, tbuf, N, H);
        gcn_scatter<<<(unsigned)cdivl((long)E * (H >> 5), TB), TB, 0, stream>>>(
            xw, srcIdx, dstIdx, dinv, tbuf, E, H);
        relu_copy<<<(unsigned)cdivl(NH, TB), TB, 0, stream>>>(tbuf, hout, NH);
    };

    // ---- position branch ----
    gcn_layer(x,   FIN, wb_gcn1, gcn1_b, pos);
    gcn_layer(pos, H,   wb_gcn2, gcn2_b, pos);

    // ---- text branch: embed -> GRU -> sum over time ----
    fill_f32<<<(unsigned)cdivl(NH, TB), TB, 0, stream>>>(hstate, 0.0f, NH);
    fill_f32<<<(unsigned)cdivl(NH, TB), TB, 0, stream>>>(textsum, 0.0f, NH);
    for (int t = 0; t < T; ++t) {
        // gi = embed[xtext[:,t]] @ Wih^T + bih
        gemm_gather1(N, ETXT, embed, xtext + t, T, wb_wih, gru_bih, gi, H3);
        // gh = h @ Whh^T + bhh
        gemm_f32(N, H, H3, hstate, H, wb_whh, gru_bhh, 0, gh, nullptr, H3);
        gru_gate<<<(unsigned)cdivl(NH, TB), TB, 0, stream>>>(gi, gh, hstate, textsum, N, H);
    }
    gcn_layer(textsum, H, wb_gcnt1, gcnt1_b, text);
    gcn_layer(text,    H, wb_gcnt2, gcnt2_b, text);

    // ---- edge features: gather-concat + linear + relu, packed bf16 into ef[E,384] ----
    gemm_gather2(E, 2 * H,    pos,            H,    wb_pos,  lin_pos_b,  ef + 0,     H3);
    gemm_gather2(E, 2 * H,    text,           H,    wb_text, lin_text_b, ef + H,     H3);
    gemm_gather2(E, 2 * IMGW, image_features, IMGW, wb_img,  lin_img_b,  ef + 2 * H, H3);

    // ---- row head (TDM-fed GEMM over bf16 ef) ----
    gemm_tdm(E, H3, ef, wb_row1, lin_row_b1, hid, H);
    head_kernel<<<(unsigned)cdivl(E, TB), TB, 0, stream>>>(hid, lin_row_W2, lin_row_b2, out_row, E, H);

    // ---- col head ----
    gemm_tdm(E, H3, ef, wb_col1, lin_col_b1, hid, H);
    head_kernel<<<(unsigned)cdivl(E, TB), TB, 0, stream>>>(hid, lin_col_W2, lin_col_b2, out_col, E, H);

    (void)n_in; (void)out_size; (void)ws_size;
}
